// DistanceBCELoss_40492951667180
// MI455X (gfx1250) — compile-verified, matrix operand order checked by reading, not run on hardware
//
#include <hip/hip_runtime.h>
#include <math.h>

#define BIGF 1.0e12f
#define NPIX (8 * 256 * 256)          // b * x * y
#define NTOT (8 * 2 * 256 * 256)      // elements in the mean

typedef __attribute__((ext_vector_type(2)))  float    v2f;
typedef __attribute__((ext_vector_type(8)))  float    v8f;
typedef __attribute__((ext_vector_type(16))) _Float16 v16h;

// --- zero the scalar output (harness poisons d_out before timing) ---------
__global__ void dbce_zero_out(float* out) {
    if (threadIdx.x == 0 && blockIdx.x == 0) out[0] = 0.0f;
}

// --- binarize target -> squared-dist seed: mask>0.5 ? BIG : 0 -------------
__global__ void dbce_prep(const int* __restrict__ target, float* __restrict__ f) {
    int i = blockIdx.x * blockDim.x + threadIdx.x;
    if (i < NPIX) f[i] = (target[i] > 0) ? BIGF : 0.0f;
}

// --- one 1-D min-plus transform, in-place, one workgroup per line ---------
// lineId = blockIdx.x in [0,2048): (b = lineId>>8, inner = lineId&255)
//   pass along y: inner_mult=256, stride=1
//   pass along x: inner_mult=1,   stride=256
__global__ void dbce_edt_line(float* __restrict__ f, int inner_mult, int stride) {
    __shared__ float line[256];
    const int lineId = blockIdx.x;
    const int base   = (lineId >> 8) * 65536 + (lineId & 255) * inner_mult;
    const int t      = threadIdx.x;

    if (stride != 1) {
        // strided column walk: hint the prefetcher (global_prefetch_b8)
        __builtin_prefetch(f + base + t * stride, 0, 0);
    }
    line[t] = f[base + t * stride];
    __syncthreads();

    float m = line[t];                       // j == t term: f[t] + 0
#pragma unroll 8
    for (int j = 0; j < 256; ++j) {
        float d = (float)(t - j);
        m = fminf(m, fmaf(d, d, line[j]));
    }
    // each block owns its line exclusively; LDS reads are done, write back
    f[base + t * stride] = m;
}

// --- wave32 sum via WMMA: D = A(16x4) x ones(4x16); lane partial in A[:,0/2]
__device__ __forceinline__ float wave_sum_wmma(float acc) {
#if __has_builtin(__builtin_amdgcn_wmma_f32_16x16x4_f32)
    v2f a;  a.x = acc;  a.y = 0.0f;          // lanes 0-15: K=0 ; lanes 16-31: K=2
    v2f b;  b.x = 1.0f; b.y = 1.0f;          // all-ones B (layout independent)
    v8f c = {};
    c = __builtin_amdgcn_wmma_f32_16x16x4_f32(
            /*neg_a=*/false, a, /*neg_b=*/false, b,
            /*c_mod=*/(short)0, c, /*reuse_a=*/false, /*reuse_b=*/false);
#else
    // fallback: f16 WMMA, same reduction algebra (D[m][n] = acc_m + acc_{m+16})
    v16h a = {};  a[0] = (_Float16)acc;      // A[m][0] (lanes 0-15), A[m][8] (16-31)
    v16h b;
    for (int i = 0; i < 16; ++i) b[i] = (_Float16)1.0f;
    v8f c = {};
    c = __builtin_amdgcn_wmma_f32_16x16x32_f16(
            false, a, false, b, (short)0, c, false, false);
#endif
    // lane L<16 holds D rows 0..7 (col L); lane L>=16 holds rows 8..15.
    float local = c[0] + c[1] + c[2] + c[3] + c[4] + c[5] + c[6] + c[7];
    float other = __shfl_xor(local, 16, 32);
    return local + other;                    // total across all 32 lanes
}

// --- fused: dist = sqrt(min(d2,BIG))+1 ; BCE for both channels ; reduce ----
__global__ void dbce_reduce(const float* __restrict__ net,
                            const int*   __restrict__ target,
                            const float* __restrict__ d2,
                            float*       __restrict__ out) {
    float acc = 0.0f;
    for (int p = blockIdx.x * blockDim.x + threadIdx.x; p < NPIX;
         p += gridDim.x * blockDim.x) {
        const int   b    = p >> 16;          // 256*256 pixels per batch
        const int   xy   = p & 65535;
        const float dist = sqrtf(fminf(d2[p], BIGF)) + 1.0f;
        const int   t    = target[p];        // 0 or 1 -> one-hot over c=2
        const int   base = (b * 2) * 65536 + xy;
        const float x0   = net[base];
        const float x1   = net[base + 65536];
        const float y0   = (t == 0) ? 1.0f : 0.0f;
        const float y1   = (t == 1) ? 1.0f : 0.0f;
        const float bce0 = fmaxf(x0, 0.0f) - x0 * y0 + log1pf(expf(-fabsf(x0)));
        const float bce1 = fmaxf(x1, 0.0f) - x1 * y1 + log1pf(expf(-fabsf(x1)));
        acc += dist * (bce0 + bce1);
    }

    // all lanes converge here -> EXEC all-ones for the WMMA
    const float wave_total = wave_sum_wmma(acc);

    __shared__ float wsum[8];                // 256 threads = 8 wave32s
    const int lane = threadIdx.x & 31;
    const int wave = threadIdx.x >> 5;
    if (lane == 0) wsum[wave] = wave_total;
    __syncthreads();
    if (threadIdx.x == 0) {
        float s = 0.0f;
        for (int w = 0; w < 8; ++w) s += wsum[w];
        atomicAdd(out, s * (1.0f / (float)NTOT));   // mean
    }
}

extern "C" void kernel_launch(void* const* d_in, const int* in_sizes, int n_in,
                              void* d_out, int out_size, void* d_ws, size_t ws_size,
                              hipStream_t stream) {
    const float* net    = (const float*)d_in[0];   // [8,2,256,256] f32
    const int*   target = (const int*)  d_in[1];   // [8,1,256,256] i32
    float*       out    = (float*)d_out;           // scalar f32
    float*       f      = (float*)d_ws;            // [8,256,256] squared-dist, 2 MB

    dbce_zero_out<<<1, 32, 0, stream>>>(out);
    dbce_prep<<<NPIX / 256, 256, 0, stream>>>(target, f);
    // pass along y (contiguous lines), then along x (stride-256 columns)
    dbce_edt_line<<<2048, 256, 0, stream>>>(f, /*inner_mult=*/256, /*stride=*/1);
    dbce_edt_line<<<2048, 256, 0, stream>>>(f, /*inner_mult=*/1,   /*stride=*/256);
    dbce_reduce<<<1024, 256, 0, stream>>>(net, target, f, out);
}
